// ViTSelfAttention_39178691674576
// MI455X (gfx1250) — matrix operand order repaired and checked by reading, not compiled, not run on hardware
//
#include <hip/hip_runtime.h>
#include <hip/hip_bf16.h>

typedef __attribute__((ext_vector_type(16))) __bf16 v16bf;
typedef __attribute__((ext_vector_type(8)))  __bf16 v8bf;
typedef __attribute__((ext_vector_type(8)))  float  v8f;

#define DEV static __device__ __forceinline__

constexpr int Bn     = 32;
constexpr int S      = 577;
constexpr int D      = 768;
constexpr int H      = 12;
constexpr int Dh     = 64;
constexpr int Spad   = 640;            // 20 * 32 -> even number of 32-key steps
constexpr int NT     = Bn * S;         // 18464 tokens = 1154 * 16 exactly
constexpr int MT     = NT / 16;        // 1154 row tiles
constexpr int STRIPS = D / Dh;         // 12 column strips of 64
constexpr int QTILES = 37;             // ceil(577/16)

// Build a 16-element bf16 fragment from two contiguous 8-half (16B) chunks.
DEV v16bf ld2(const __bf16* p0, const __bf16* p1) {
  v16bf r;
  v8bf a = *reinterpret_cast<const v8bf*>(p0);
  v8bf b = *reinterpret_cast<const v8bf*>(p1);
#pragma unroll
  for (int i = 0; i < 8; ++i) { r[i] = a[i]; r[i + 8] = b[i]; }
  return r;
}

DEV void ds_fence() {
#if __has_builtin(__builtin_amdgcn_s_wait_dscnt)
  __builtin_amdgcn_s_wait_dscnt(0);
#else
  asm volatile("" ::: "memory");   // per-wave DS ops are architecturally in-order
#endif
}

// Hard scheduling fence: keep the software-pipeline stage boundaries intact
// (stops the machine scheduler from sinking prefetch loads down to their uses).
DEV void sched_fence() {
#if __has_builtin(__builtin_amdgcn_sched_barrier)
  __builtin_amdgcn_sched_barrier(0);
#endif
}

DEV v8f wmma_bf16(v16bf a, v16bf b, v8f c) {
  return __builtin_amdgcn_wmma_f32_16x16x32_bf16(false, a, false, b,
                                                 (short)0, c, false, false);
}

// ---------------------------------------------------------------------------
// Utility kernels
// ---------------------------------------------------------------------------
__global__ void __launch_bounds__(256) zero_ws16(uint4* __restrict__ p, long n16) {
  long i = (long)blockIdx.x * blockDim.x + threadIdx.x;
  if (i < n16) p[i] = make_uint4(0u, 0u, 0u, 0u);
}

__global__ void __launch_bounds__(256) f32_to_bf16(const float* __restrict__ src,
                                                   __bf16* __restrict__ dst, int n) {
  int i = blockIdx.x * blockDim.x + threadIdx.x;
  if (i < n) dst[i] = (__bf16)src[i];
}

// ---------------------------------------------------------------------------
// QKV projection: out = x @ W^T + b  (NT x D) in bf16, split-head layouts:
//   Q, K -> [B, H, Spad, Dh] row-major
//   V    -> [B, H, Dh, Spad] (transposed so P·V B-fragments are contiguous)
// One wave computes a 16x64 strip; fully-unrolled 24-step K-loop, software
// pipelined with sched_barriers so step k+1 loads stay ahead of step k WMMAs.
// ---------------------------------------------------------------------------
__global__ void __launch_bounds__(256) qkv_gemm(
    const __bf16* __restrict__ X, const __bf16* __restrict__ W,
    const float* __restrict__ bq, const float* __restrict__ bk,
    const float* __restrict__ bv,
    __bf16* __restrict__ Q, __bf16* __restrict__ Kd, __bf16* __restrict__ V) {
  const int lane  = threadIdx.x & 31;
  const int wv    = threadIdx.x >> 5;
  const int gw    = blockIdx.x * 8 + wv;
  const int mat   = gw / (MT * STRIPS);         // 0=Q 1=K 2=V
  const int rem   = gw % (MT * STRIPS);
  const int mtile = rem / STRIPS;
  const int strip = rem % STRIPS;               // == head index of this strip
  const int ln    = lane & 15;
  const int hi    = lane >> 4;

  const __bf16* Wm   = W + (size_t)mat * D * D;
  const float*  bias = (mat == 0) ? bq : (mat == 1) ? bk : bv;

  const __bf16* xrow = X + (size_t)(mtile * 16 + ln) * D;  // A: lane = M row

  v8f acc[4];
#pragma unroll
  for (int t = 0; t < 4; ++t)
#pragma unroll
    for (int i = 0; i < 8; ++i) acc[t][i] = 0.0f;

  // -- prologue: fragments for k0 = 0
  v16bf a_cur = ld2(xrow + hi * 16, xrow + hi * 16 + 8);
  v16bf b_cur[4];
#pragma unroll
  for (int t = 0; t < 4; ++t) {
    const __bf16* w = Wm + (size_t)(strip * 64 + t * 16 + ln) * D + hi * 16;
    b_cur[t] = ld2(w, w + 8);
  }

  // Fully unrolled + pipelined main loop; sched_fence() pins the prefetch
  // loads ahead of the current step's WMMAs in the final schedule.
#pragma unroll
  for (int k0 = 0; k0 < D; k0 += 32) {
    const int kn = k0 + 32;
    // Prefetch next step's fragments first (dead on final step -> DCE'd;
    // any 64B over-read past the last row stays inside the workspace).
    v16bf a_nxt = ld2(xrow + kn + hi * 16, xrow + kn + hi * 16 + 8);
    v16bf b_nxt[4];
#pragma unroll
    for (int t = 0; t < 4; ++t) {
      const __bf16* w = Wm + (size_t)(strip * 64 + t * 16 + ln) * D + kn + hi * 16;
      b_nxt[t] = ld2(w, w + 8);
    }
    sched_fence();   // loads above stay above; WMMAs below wait only on OLD loads
#pragma unroll
    for (int t = 0; t < 4; ++t) acc[t] = wmma_bf16(a_cur, b_cur[t], acc[t]);
    sched_fence();
    a_cur = a_nxt;
#pragma unroll
    for (int t = 0; t < 4; ++t) b_cur[t] = b_nxt[t];
  }

  // Epilogue: bias add, bf16 convert, scatter into head layouts.
#pragma unroll
  for (int r = 0; r < 8; ++r) {
    const int tok = mtile * 16 + r + 8 * hi;   // C: VGPR r -> M = r + 8*hi
    const int bI  = tok / S;
    const int sI  = tok - bI * S;
#pragma unroll
    for (int t = 0; t < 4; ++t) {
      const int dd = t * 16 + ln;              // C: lane -> N
      const float v = acc[t][r] + bias[strip * 64 + dd];
      if (mat == 2) {
        V[(((size_t)bI * H + strip) * Dh + dd) * Spad + sI] = (__bf16)v;
      } else {
        __bf16* dst = (mat == 0) ? Q : Kd;
        dst[(((size_t)bI * H + strip) * Spad + sI) * Dh + dd] = (__bf16)v;
      }
    }
  }
}

// ---------------------------------------------------------------------------
// Flash attention: one wave per (b, h, 16-row q-tile); 20 steps of 32 keys,
// run as 10 double-steps with explicit kf ping-pong. Per sub-step: score
// WMMAs on preloaded K frags; V frags + next-step K frags issued (and pinned
// by sched_fence) before the softmax VALU work that hides their latency;
// context WMMAs consume V at the end.
// ---------------------------------------------------------------------------
__global__ void __launch_bounds__(256) attn(
    const __bf16* __restrict__ Q, const __bf16* __restrict__ Kd,
    const __bf16* __restrict__ V, float* __restrict__ out) {
  __shared__ __bf16 plds[8][16][32];

  const int lane = threadIdx.x & 31;
  const int wv   = threadIdx.x >> 5;
  const int gw   = blockIdx.x * 8 + wv;
  const int qt   = gw % QTILES;
  const int h    = (gw / QTILES) % H;
  const int bI   = gw / (QTILES * H);
  const int ln   = lane & 15;
  const int hi   = lane >> 4;

  const __bf16* Qh = Q  + ((size_t)bI * H + h) * Spad * Dh;
  const __bf16* Kh = Kd + ((size_t)bI * H + h) * Spad * Dh;
  const __bf16* Vh = V  + ((size_t)bI * H + h) * Dh * Spad;

  // Preload Q A-fragments for both 32-wide d-chunks.
  const int qrow = qt * 16 + ln;
  v16bf qf[2];
#pragma unroll
  for (int c = 0; c < 2; ++c) {
    const __bf16* qp = Qh + (size_t)qrow * Dh + c * 32;
    qf[c] = ld2(qp + hi * 16, qp + hi * 16 + 8);
  }

  v8f acc[4];
  float mrow[8], lrow[8];
#pragma unroll
  for (int t = 0; t < 4; ++t)
#pragma unroll
    for (int i = 0; i < 8; ++i) acc[t][i] = 0.0f;
#pragma unroll
  for (int r = 0; r < 8; ++r) { mrow[r] = -1e30f; lrow[r] = 0.0f; }

  // One 32-key sub-step. kc = K fragments for this step (consumed),
  // kn = buffer that receives the prefetch for the NEXT step.
  auto step = [&](int k0, v16bf (&kc)[2][2], v16bf (&kn)[2][2]) {
    // ---- scores on preloaded K fragments
    v8f sc[2];
#pragma unroll
    for (int t = 0; t < 2; ++t) {
#pragma unroll
      for (int i = 0; i < 8; ++i) sc[t][i] = 0.0f;
#pragma unroll
      for (int c = 0; c < 2; ++c) sc[t] = wmma_bf16(qf[c], kc[t][c], sc[t]);
    }
    sched_fence();

    // ---- issue this step's V fragments (consumed after softmax)
    v16bf vf[4];
#pragma unroll
    for (int t = 0; t < 4; ++t) {
      const __bf16* vp = Vh + (size_t)(t * 16 + ln) * Spad + k0 + hi * 16;
      vf[t] = ld2(vp, vp + 8);
    }
    // ---- prefetch next step's K fragments (dead past the last step;
    //      over-read stays inside the workspace)
#pragma unroll
    for (int t = 0; t < 2; ++t)
#pragma unroll
      for (int c = 0; c < 2; ++c) {
        const __bf16* kp =
            Kh + (size_t)(k0 + 32 + t * 16 + ln) * Dh + c * 32 + hi * 16;
        kn[t][c] = ld2(kp, kp + 8);
      }
    sched_fence();   // V/K loads stay issued here; softmax below hides them

    // ---- scale + key mask
    const int kidx0 = k0 + ln, kidx1 = k0 + 16 + ln;
#pragma unroll
    for (int r = 0; r < 8; ++r) {
      sc[0][r] = (kidx0 < S) ? sc[0][r] * 0.125f : -1e30f;
      sc[1][r] = (kidx1 < S) ? sc[1][r] * 0.125f : -1e30f;
    }

    // ---- online softmax per row (rows live across 16-lane halves)
#pragma unroll
    for (int r = 0; r < 8; ++r) {
      float mx = fmaxf(sc[0][r], sc[1][r]);
#pragma unroll
      for (int msk = 1; msk < 16; msk <<= 1) mx = fmaxf(mx, __shfl_xor(mx, msk, 32));
      const float mn   = fmaxf(mrow[r], mx);
      const float corr = __expf(mrow[r] - mn);
      const float p0   = __expf(sc[0][r] - mn);
      const float p1   = __expf(sc[1][r] - mn);
      float rs = p0 + p1;
#pragma unroll
      for (int msk = 1; msk < 16; msk <<= 1) rs += __shfl_xor(rs, msk, 32);
      lrow[r] = lrow[r] * corr + rs;
      mrow[r] = mn;
#pragma unroll
      for (int t = 0; t < 4; ++t) acc[t][r] *= corr;
      const int m = r + 8 * hi;                 // C layout -> matrix row
      plds[wv][m][ln]      = (__bf16)p0;
      plds[wv][m][ln + 16] = (__bf16)p1;
    }
    ds_fence();

    // ---- P as A-fragment (16x32) from LDS
    const __bf16* pr = &plds[wv][ln][0];        // A: lane = M row
    v16bf pf = ld2(pr + hi * 16, pr + hi * 16 + 8);

    // ---- context: acc[t] += P · V  (V transposed: rows = d, cols = keys)
#pragma unroll
    for (int t = 0; t < 4; ++t) acc[t] = wmma_bf16(pf, vf[t], acc[t]);
  };

  // -- prologue: K fragments for k0 = 0  ([key-subtile t][d-chunk c])
  v16bf kfA[2][2], kfB[2][2];
#pragma unroll
  for (int t = 0; t < 2; ++t)
#pragma unroll
    for (int c = 0; c < 2; ++c) {
      const __bf16* kp = Kh + (size_t)(t * 16 + ln) * Dh + c * 32 + hi * 16;
      kfA[t][c] = ld2(kp, kp + 8);
    }

  for (int k0 = 0; k0 < Spad; k0 += 64) {   // 10 double-steps
    step(k0,      kfA, kfB);
    step(k0 + 32, kfB, kfA);
  }

  // ---- normalize and store fp32 output [B, S, H*Dh]
#pragma unroll
  for (int r = 0; r < 8; ++r) {
    const int srow = qt * 16 + r + 8 * hi;
    if (srow < S) {
      const float inv = 1.0f / lrow[r];
#pragma unroll
      for (int t = 0; t < 4; ++t)
        out[((size_t)bI * S + srow) * D + h * Dh + t * 16 + ln] = acc[t][r] * inv;
    }
  }
}

// ---------------------------------------------------------------------------
// Host launcher
// ---------------------------------------------------------------------------
extern "C" void kernel_launch(void* const* d_in, const int* in_sizes, int n_in,
                              void* d_out, int out_size, void* d_ws, size_t ws_size,
                              hipStream_t stream) {
  const float* x  = (const float*)d_in[0];
  const float* Wq = (const float*)d_in[1];
  const float* bq = (const float*)d_in[2];
  const float* Wk = (const float*)d_in[3];
  const float* bk = (const float*)d_in[4];
  const float* Wv = (const float*)d_in[5];
  const float* bv = (const float*)d_in[6];
  float* out = (float*)d_out;

  char* ws = (char*)d_ws;
  size_t off = 0;
  __bf16* Xb = (__bf16*)(ws + off); off += (size_t)NT * D * 2;        // 28.4 MB
  __bf16* Wb = (__bf16*)(ws + off); off += (size_t)3 * D * D * 2;     //  3.5 MB
  const size_t hsz = (size_t)Bn * H * Spad * Dh * 2;                  // 31.5 MB each
  __bf16* Qb = (__bf16*)(ws + off); off += hsz;
  __bf16* Kb = (__bf16*)(ws + off); off += hsz;
  __bf16* Vb = (__bf16*)(ws + off); off += hsz;

  // Zero padded Q/K/V region (pad rows must be exact 0, not 0xAA poison).
  const long n16 = (long)(3 * hsz / 16);
  zero_ws16<<<(int)((n16 + 255) / 256), 256, 0, stream>>>((uint4*)Qb, n16);

  // fp32 -> bf16 staging.
  f32_to_bf16<<<(NT * D + 255) / 256, 256, 0, stream>>>(x, Xb, NT * D);
  f32_to_bf16<<<(D * D + 255) / 256, 256, 0, stream>>>(Wq, Wb,             D * D);
  f32_to_bf16<<<(D * D + 255) / 256, 256, 0, stream>>>(Wk, Wb + D * D,     D * D);
  f32_to_bf16<<<(D * D + 255) / 256, 256, 0, stream>>>(Wv, Wb + 2 * D * D, D * D);

  // QKV projection: 3 * 1154 * 12 waves, 8 waves per block (exact).
  qkv_gemm<<<(3 * MT * STRIPS) / 8, 256, 0, stream>>>(Xb, Wb, bq, bk, bv, Qb, Kb, Vb);

  // Attention: 32*12*37 = 14208 waves, 8 per block (exact).
  attn<<<(Bn * H * QTILES) / 8, 256, 0, stream>>>(Qb, Kb, Vb, out);
}
